// PropagationRWKV_90993177133308
// MI455X (gfx1250) — compile-verified, matrix operand order checked by reading, not run on hardware
//
#include <hip/hip_runtime.h>
#include <math.h>

#define CDIM 256
#define NSEQ 16384
#define NCHUNK 64
#define LCHUNK (NSEQ / NCHUNK)   // 256
#define LDS_STRIDE 264           // pad 256 -> 264 half-words to avoid LDS bank conflicts
#define LN_EPS 1e-5f

typedef __attribute__((ext_vector_type(16))) __bf16 v16bf;
typedef __attribute__((ext_vector_type(8)))  float v8f;

union Frag16 {
  v16bf v;
  uint4 q[2];
};

__device__ __forceinline__ unsigned short f32_bf16(float f) {
  union { float f; unsigned int u; } cv;
  cv.f = f;
  unsigned int u = cv.u;
  u += 0x7FFFu + ((u >> 16) & 1u);   // round-to-nearest-even
  return (unsigned short)(u >> 16);
}

__device__ __forceinline__ float bf16_f32(unsigned short s) {
  union { unsigned int u; float f; } cv;
  cv.u = ((unsigned int)s) << 16;
  return cv.f;
}

__device__ __forceinline__ v8f zero8() {
  v8f z;
  #pragma unroll
  for (int i = 0; i < 8; ++i) z[i] = 0.0f;
  return z;
}

__device__ __forceinline__ Frag16 load_frag(const unsigned short* p) {
  // 16 bf16 = two 16-byte contiguous halves (K..K+7 and K+16..K+23 per lane-half)
  Frag16 f;
  f.q[0] = *reinterpret_cast<const uint4*>(p);
  f.q[1] = *reinterpret_cast<const uint4*>(p + 16);
  return f;
}

__device__ __forceinline__ v8f wmma_bf16(const Frag16& a, const Frag16& b, v8f c) {
  return __builtin_amdgcn_wmma_f32_16x16x32_bf16(false, a.v, false, b.v, (short)0, c,
                                                 false, false);
}

// ---------------------------------------------------------------------------
// Kernel 0: weights f32 -> bf16, decay vectors
// ---------------------------------------------------------------------------
__global__ void __launch_bounds__(256) rwkv_prep_kernel(
    const float* __restrict__ Wk, const float* __restrict__ Wv,
    const float* __restrict__ Wr, const float* __restrict__ Wo,
    const float* __restrict__ td,
    unsigned short* __restrict__ wk, unsigned short* __restrict__ wv,
    unsigned short* __restrict__ wr, unsigned short* __restrict__ wo,
    float* __restrict__ decay, float* __restrict__ decayL) {
  const int i = blockIdx.x * 256 + threadIdx.x;          // 0 .. 4*65536-1
  const int m = i >> 16;
  const int j = i & 0xFFFF;
  const float* src = (m == 0) ? Wk : (m == 1) ? Wv : (m == 2) ? Wr : Wo;
  unsigned short* dst = (m == 0) ? wk : (m == 1) ? wv : (m == 2) ? wr : wo;
  dst[j] = f32_bf16(src[j]);
  if (i < CDIM) {
    const float t = fmaxf(td[i], 0.0f);
    decay[i]  = __expf(-t);                              // exp(-relu(td))
    decayL[i] = __expf(-(float)LCHUNK * t);              // decay^LCHUNK exactly
  }
}

// ---------------------------------------------------------------------------
// Kernel 1: fused LayerNorm + 3 WMMA GEMMs (k f32, v bf16, r=sigmoid bf16)
// Block: 256 threads = 8 waves, 16 rows of x per block.
// ---------------------------------------------------------------------------
__global__ void __launch_bounds__(256) rwkv_kvr_kernel(
    const float* __restrict__ x, const float* __restrict__ lnw, const float* __restrict__ lnb,
    const unsigned short* __restrict__ wk, const unsigned short* __restrict__ wv,
    const unsigned short* __restrict__ wr,
    float* __restrict__ kout, unsigned short* __restrict__ vout,
    unsigned short* __restrict__ rout) {
  __shared__ unsigned short xn[16 * LDS_STRIDE];
  const int tid  = threadIdx.x;
  const int lane = tid & 31;
  const int wave = tid >> 5;
  const int mrow = lane & 15;
  const int half = (lane >> 4) << 3;     // 0 or 8 : K-half per lane group
  const int rowbase = blockIdx.x * 16;

  // ---- LayerNorm: each wave handles 2 rows; lane covers 8 channels ----
  #pragma unroll
  for (int rr = 0; rr < 2; ++rr) {
    const int r = wave * 2 + rr;
    const float* xr = x + (size_t)(rowbase + r) * CDIM + lane * 8;
    float4 a = *reinterpret_cast<const float4*>(xr);
    float4 b = *reinterpret_cast<const float4*>(xr + 4);
    float s  = a.x + a.y + a.z + a.w + b.x + b.y + b.z + b.w;
    float s2 = a.x*a.x + a.y*a.y + a.z*a.z + a.w*a.w
             + b.x*b.x + b.y*b.y + b.z*b.z + b.w*b.w;
    #pragma unroll
    for (int off = 16; off > 0; off >>= 1) {   // wave32 reduction
      s  += __shfl_xor(s, off);
      s2 += __shfl_xor(s2, off);
    }
    const float mu   = s * (1.0f / CDIM);
    const float rstd = rsqrtf(s2 * (1.0f / CDIM) - mu * mu + LN_EPS);
    const int c0 = lane * 8;
    float4 w0 = *reinterpret_cast<const float4*>(lnw + c0);
    float4 w1 = *reinterpret_cast<const float4*>(lnw + c0 + 4);
    float4 g0 = *reinterpret_cast<const float4*>(lnb + c0);
    float4 g1 = *reinterpret_cast<const float4*>(lnb + c0 + 4);
    unsigned int h0 = (unsigned)f32_bf16((a.x - mu) * rstd * w0.x + g0.x)
                    | ((unsigned)f32_bf16((a.y - mu) * rstd * w0.y + g0.y) << 16);
    unsigned int h1 = (unsigned)f32_bf16((a.z - mu) * rstd * w0.z + g0.z)
                    | ((unsigned)f32_bf16((a.w - mu) * rstd * w0.w + g0.w) << 16);
    unsigned int h2 = (unsigned)f32_bf16((b.x - mu) * rstd * w1.x + g1.x)
                    | ((unsigned)f32_bf16((b.y - mu) * rstd * w1.y + g1.y) << 16);
    unsigned int h3 = (unsigned)f32_bf16((b.z - mu) * rstd * w1.z + g1.z)
                    | ((unsigned)f32_bf16((b.w - mu) * rstd * w1.w + g1.w) << 16);
    uint4 pk; pk.x = h0; pk.y = h1; pk.z = h2; pk.w = h3;
    *reinterpret_cast<uint4*>(&xn[r * LDS_STRIDE + c0]) = pk;
  }
  __syncthreads();

  // ---- GEMM: wave owns n in [wave*32, wave*32+32) for all three matrices ----
  const int n0 = wave * 32;
  v8f acc[3][2];
  #pragma unroll
  for (int m = 0; m < 3; ++m)
    #pragma unroll
    for (int t = 0; t < 2; ++t) acc[m][t] = zero8();

  for (int k0 = 0; k0 < CDIM; k0 += 32) {
    Frag16 afrag;
    {
      const unsigned short* p = &xn[mrow * LDS_STRIDE + k0 + half];
      afrag.q[0] = *reinterpret_cast<const uint4*>(p);
      afrag.q[1] = *reinterpret_cast<const uint4*>(p + 16);
    }
    #pragma unroll
    for (int t = 0; t < 2; ++t) {
      const int n = n0 + t * 16 + mrow;
      const int boff = n * CDIM + k0 + half;
      Frag16 bk = load_frag(wk + boff);
      Frag16 bv = load_frag(wv + boff);
      Frag16 br = load_frag(wr + boff);
      acc[0][t] = wmma_bf16(afrag, bk, acc[0][t]);
      acc[1][t] = wmma_bf16(afrag, bv, acc[1][t]);
      acc[2][t] = wmma_bf16(afrag, br, acc[2][t]);
    }
  }

  const int hi = lane >> 4;
  #pragma unroll
  for (int t = 0; t < 2; ++t) {
    const int col = n0 + t * 16 + mrow;
    #pragma unroll
    for (int j = 0; j < 8; ++j) {
      const size_t idx = (size_t)(rowbase + j + 8 * hi) * CDIM + col;
      kout[idx] = acc[0][t][j];
      vout[idx] = f32_bf16(acc[1][t][j]);
      const float rv = acc[2][t][j];
      rout[idx] = f32_bf16(1.0f / (1.0f + __expf(-rv)));
    }
  }
}

// ---------------------------------------------------------------------------
// Kernel 2: per-chunk local decay scan (in place over kbuf), emit chunk sums
// ---------------------------------------------------------------------------
__global__ void __launch_bounds__(256) rwkv_scan_local_kernel(
    float* __restrict__ kbuf, const float* __restrict__ decay,
    float* __restrict__ carries) {
  const int c = threadIdx.x;
  const int chunk = blockIdx.x;
  const int b = blockIdx.y;
  const float d = decay[c];
  size_t idx = ((size_t)b * NSEQ + (size_t)chunk * LCHUNK) * CDIM + c;
  float s = 0.0f;
  for (int t = 0; t < LCHUNK; ++t) {
    s = fmaf(s, d, kbuf[idx]);
    kbuf[idx] = s;
    idx += CDIM;
  }
  carries[((size_t)b * NCHUNK + chunk) * CDIM + c] = s;
}

// ---------------------------------------------------------------------------
// Kernel 3: exclusive prefix over chunk carries (64 steps per (b,c))
// ---------------------------------------------------------------------------
__global__ void __launch_bounds__(256) rwkv_carry_kernel(
    float* __restrict__ carries, const float* __restrict__ decayL) {
  const int c = threadIdx.x;
  const int b = blockIdx.x;
  const float dl = decayL[c];
  float p = 0.0f;
  for (int i = 0; i < NCHUNK; ++i) {
    const size_t idx = ((size_t)b * NCHUNK + i) * CDIM + c;
    const float t = carries[idx];
    carries[idx] = p;
    p = fmaf(p, dl, t);
  }
}

// ---------------------------------------------------------------------------
// Kernel 4: apply carries + y = r*(state+v), write y as bf16 in place over r
// ---------------------------------------------------------------------------
__global__ void __launch_bounds__(256) rwkv_apply_kernel(
    const float* __restrict__ kbuf, const unsigned short* __restrict__ vbuf,
    unsigned short* __restrict__ rbuf,
    const float* __restrict__ decay, const float* __restrict__ carries) {
  const int c = threadIdx.x;
  const int chunk = blockIdx.x;
  const int b = blockIdx.y;
  const float d  = decay[c];
  const float S0 = carries[((size_t)b * NCHUNK + chunk) * CDIM + c];
  size_t idx = ((size_t)b * NSEQ + (size_t)chunk * LCHUNK) * CDIM + c;
  float dpow = d;
  for (int t = 0; t < LCHUNK; ++t) {
    const float state = fmaf(S0, dpow, kbuf[idx]);
    dpow *= d;
    const float v = bf16_f32(vbuf[idx]);
    const float r = bf16_f32(rbuf[idx]);
    rbuf[idx] = f32_bf16(r * (state + v));
    idx += CDIM;
  }
}

// ---------------------------------------------------------------------------
// Kernel 5: out = y @ Wo^T via bf16 WMMA, f32 output
// ---------------------------------------------------------------------------
__global__ void __launch_bounds__(256) rwkv_out_kernel(
    const unsigned short* __restrict__ y, const unsigned short* __restrict__ wo,
    float* __restrict__ out) {
  const int tid  = threadIdx.x;
  const int lane = tid & 31;
  const int wave = tid >> 5;
  const int mrow = lane & 15;
  const int half = (lane >> 4) << 3;
  const int rowbase = blockIdx.x * 16;
  const int n0 = wave * 32;
  v8f acc0 = zero8(), acc1 = zero8();
  const unsigned short* arow = y + (size_t)(rowbase + mrow) * CDIM;
  for (int k0 = 0; k0 < CDIM; k0 += 32) {
    Frag16 a = load_frag(arow + k0 + half);
    Frag16 b0 = load_frag(wo + (n0 + mrow) * CDIM + k0 + half);
    Frag16 b1 = load_frag(wo + (n0 + 16 + mrow) * CDIM + k0 + half);
    acc0 = wmma_bf16(a, b0, acc0);
    acc1 = wmma_bf16(a, b1, acc1);
  }
  const int hi = lane >> 4;
  #pragma unroll
  for (int j = 0; j < 8; ++j) {
    const size_t row = (size_t)(rowbase + j + 8 * hi) * CDIM;
    out[row + n0 + mrow]      = acc0[j];
    out[row + n0 + 16 + mrow] = acc1[j];
  }
}

// ---------------------------------------------------------------------------
extern "C" void kernel_launch(void* const* d_in, const int* in_sizes, int n_in,
                              void* d_out, int out_size, void* d_ws, size_t ws_size,
                              hipStream_t stream) {
  const float* x   = (const float*)d_in[0];
  const float* td  = (const float*)d_in[1];
  const float* Wk  = (const float*)d_in[2];
  const float* Wv  = (const float*)d_in[3];
  const float* Wr  = (const float*)d_in[4];
  const float* Wo  = (const float*)d_in[5];
  const float* lnw = (const float*)d_in[6];
  const float* lnb = (const float*)d_in[7];
  float* out = (float*)d_out;

  const int rows  = in_sizes[0] / CDIM;       // B*N = 131072
  const int batch = rows / NSEQ;              // 8
  const size_t elems = (size_t)rows * CDIM;   // 33.5M

  char* ws = (char*)d_ws;
  size_t off = 0;
  auto take = [&](size_t bytes) -> void* {
    void* p = ws + off;
    off += (bytes + 255) & ~(size_t)255;
    return p;
  };
  unsigned short* wk = (unsigned short*)take((size_t)CDIM * CDIM * 2);
  unsigned short* wv = (unsigned short*)take((size_t)CDIM * CDIM * 2);
  unsigned short* wr = (unsigned short*)take((size_t)CDIM * CDIM * 2);
  unsigned short* wo = (unsigned short*)take((size_t)CDIM * CDIM * 2);
  float* decay   = (float*)take(CDIM * 4);
  float* decayL  = (float*)take(CDIM * 4);
  float* carries = (float*)take((size_t)batch * NCHUNK * CDIM * 4);
  float* kbuf    = (float*)take(elems * 4);
  unsigned short* vbuf = (unsigned short*)take(elems * 2);
  unsigned short* rbuf = (unsigned short*)take(elems * 2);
  (void)n_in; (void)out_size; (void)ws_size;

  rwkv_prep_kernel<<<dim3((4 * CDIM * CDIM) / 256), dim3(256), 0, stream>>>(
      Wk, Wv, Wr, Wo, td, wk, wv, wr, wo, decay, decayL);
  rwkv_kvr_kernel<<<dim3(rows / 16), dim3(256), 0, stream>>>(
      x, lnw, lnb, wk, wv, wr, kbuf, vbuf, rbuf);
  rwkv_scan_local_kernel<<<dim3(NCHUNK, batch), dim3(CDIM), 0, stream>>>(
      kbuf, decay, carries);
  rwkv_carry_kernel<<<dim3(batch), dim3(CDIM), 0, stream>>>(carries, decayL);
  rwkv_apply_kernel<<<dim3(NCHUNK, batch), dim3(CDIM), 0, stream>>>(
      kbuf, vbuf, rbuf, decay, carries);
  rwkv_out_kernel<<<dim3(rows / 16), dim3(256), 0, stream>>>(rbuf, wo, out);
}
